// ComnetModel_71408126263501
// MI455X (gfx1250) — matrix operand order repaired and checked by reading, not compiled, not run on hardware
//
#include <hip/hip_runtime.h>
#include <hip/hip_bf16.h>

// ---------------------------------------------------------------------------
// Types for WMMA operands (wave32, 16x16x32 bf16 -> f32)
// ---------------------------------------------------------------------------
typedef __attribute__((ext_vector_type(16))) __bf16        bf16x16;
typedef __attribute__((ext_vector_type(8)))  float         f32x8;
typedef __attribute__((ext_vector_type(8)))  unsigned int  u32x8;

#define N_NODES  50000
#define N_EDGES  400000
#define HDIM     64
#define EDIM     8
#define MSG_HID  128
#define RD_HID   128

// Packed-weight tile table (each 32x16 bf16 B-tile = 32 lanes * 8 dwords = 256 u32)
#define TILE_W1   0     // W_msg1: 5 ktiles x 8 ntiles = 40  (K padded 136->160)
#define TILE_W2   40    // W_msg2: 4 x 4 = 16
#define TILE_WX   56    // W_x:    2 x 12 = 24
#define TILE_WH   80    // W_h:    2 x 12 = 24
#define TILE_WR1  104   // W_r1:   2 x 8  = 16
#define N_TILES_TOTAL 120

// Workspace layout (bytes)
#define EFBF_OFF  131072                              // packed weights rounded up
#define HBF_OFF   (EFBF_OFF + N_EDGES * EDIM * 2)     // +6.4 MB edge feats bf16
#define AGG_OFF   (HBF_OFF  + N_NODES * HDIM * 2)     // +6.4 MB node state bf16
#define HBUF_OFF  (AGG_OFF  + N_NODES * HDIM * 4)     // +12.8 MB agg f32, then h f32

// ---------------------------------------------------------------------------
// Helpers
// ---------------------------------------------------------------------------
__device__ __forceinline__ f32x8 wmma_bf16(u32x8 a, u32x8 b, f32x8 c) {
  return __builtin_amdgcn_wmma_f32_16x16x32_bf16(
      false, __builtin_bit_cast(bf16x16, a),
      false, __builtin_bit_cast(bf16x16, b),
      (short)0, c, false, false);
}
__device__ __forceinline__ f32x8 wmma_bf16v(bf16x16 a, u32x8 b, f32x8 c) {
  return __builtin_amdgcn_wmma_f32_16x16x32_bf16(
      false, a, false, __builtin_bit_cast(bf16x16, b),
      (short)0, c, false, false);
}

// Load a pre-packed 32x16 B tile: 32 bytes per lane, fully coalesced.
__device__ __forceinline__ u32x8 load_tileB(const unsigned int* p) {
  const uint4* q = (const uint4*)p;
  uint4 x = q[0], y = q[1];
  u32x8 o;
  o[0] = x.x; o[1] = x.y; o[2] = x.z; o[3] = x.w;
  o[4] = y.x; o[5] = y.y; o[6] = y.z; o[7] = y.w;
  return o;
}

// A-tile (16x32 bf16) registers straight from a bf16 row: two 16B runs.
// Lane holds row r = lane%16; half = lane/16. Cols [c0..c0+7] and [c0+16..c0+23].
__device__ __forceinline__ u32x8 loadA_bf16row(const unsigned short* rowp, int c0) {
  uint4 x = *(const uint4*)(rowp + c0);
  uint4 y = *(const uint4*)(rowp + c0 + 16);
  u32x8 o;
  o[0] = x.x; o[1] = x.y; o[2] = x.z; o[3] = x.w;
  o[4] = y.x; o[5] = y.y; o[6] = y.z; o[7] = y.w;
  return o;
}

// A-tile from an f32 row, converting with native bf16 casts.
__device__ __forceinline__ bf16x16 loadA_f32row(const float* rowp, int c0) {
  const float4* p0 = (const float4*)(rowp + c0);
  const float4* p1 = (const float4*)(rowp + c0 + 16);
  float4 a0 = p0[0], a1 = p0[1];
  float4 b0 = p1[0], b1 = p1[1];
  bf16x16 o;
  o[0]  = (__bf16)a0.x; o[1]  = (__bf16)a0.y; o[2]  = (__bf16)a0.z; o[3]  = (__bf16)a0.w;
  o[4]  = (__bf16)a1.x; o[5]  = (__bf16)a1.y; o[6]  = (__bf16)a1.z; o[7]  = (__bf16)a1.w;
  o[8]  = (__bf16)b0.x; o[9]  = (__bf16)b0.y; o[10] = (__bf16)b0.z; o[11] = (__bf16)b0.w;
  o[12] = (__bf16)b1.x; o[13] = (__bf16)b1.y; o[14] = (__bf16)b1.z; o[15] = (__bf16)b1.w;
  return o;
}

__device__ __forceinline__ unsigned short bf16_bits(float x) {
  return __builtin_bit_cast(unsigned short, (__bf16)x);
}
__device__ __forceinline__ float sigmoid_f(float x) {
  return 1.0f / (1.0f + __expf(-x));
}
__device__ __forceinline__ float tanh_f(float x) {
  return 1.0f - 2.0f / (__expf(2.0f * x) + 1.0f);
}

// ---------------------------------------------------------------------------
// Kernel 0: pack all weight matrices into WMMA B-tile layout (bf16).
// B-tile layout (16-bit, 32x16): lane L holds column n = nt*16 + L%16,
// dword v holds K = kt*32 + (L/16)*8 + 2v (+8 if v>=4) and +1.
// ---------------------------------------------------------------------------
__global__ __launch_bounds__(256) void pack_weights_kernel(
    const float* __restrict__ W1, const float* __restrict__ W2,
    const float* __restrict__ Wx, const float* __restrict__ Wh,
    const float* __restrict__ Wr1, unsigned int* __restrict__ packed) {
  int t    = blockIdx.x;
  int lane = threadIdx.x & 31;
  int v    = threadIdx.x >> 5;   // 0..7

  const float* W; int Kact, N, kt, nt;
  if (t < TILE_W2)        { int u = t;            W = W1;  Kact = 2*HDIM+EDIM; N = MSG_HID; kt = u/8;  nt = u%8;  }
  else if (t < TILE_WH)   { int u = t - TILE_W2;  W = W2;  Kact = MSG_HID;     N = HDIM;    kt = u/4;  nt = u%4;  }
  else if (t < TILE_WR1)  { int u = t - TILE_WX;  W = Wx;  Kact = HDIM;        N = 3*HDIM;  kt = u/12; nt = u%12; }
  else                    { int u = t - TILE_WR1; W = Wr1; Kact = HDIM;        N = RD_HID;  kt = u/8;  nt = u%8;  }
  // NOTE: ranges above collapsed wrongly if left as-is; use explicit chain:
  if (t < TILE_W2)        { int u = t;            W = W1;  Kact = 2*HDIM+EDIM; N = MSG_HID; kt = u/8;  nt = u%8;  }
  else if (t < TILE_WX)   { int u = t - TILE_W2;  W = W2;  Kact = MSG_HID;     N = HDIM;    kt = u/4;  nt = u%4;  }
  else if (t < TILE_WH)   { int u = t - TILE_WX;  W = Wx;  Kact = HDIM;        N = 3*HDIM;  kt = u/12; nt = u%12; }
  else if (t < TILE_WR1)  { int u = t - TILE_WH;  W = Wh;  Kact = HDIM;        N = 3*HDIM;  kt = u/12; nt = u%12; }
  else                    { int u = t - TILE_WR1; W = Wr1; Kact = HDIM;        N = RD_HID;  kt = u/8;  nt = u%8;  }

  int n  = nt * 16 + (lane & 15);
  int k0 = kt * 32 + (lane >> 4) * 8 + 2 * v + (v >= 4 ? 8 : 0);
  float f0 = (k0     < Kact) ? W[k0 * N + n]       : 0.0f;
  float f1 = (k0 + 1 < Kact) ? W[(k0 + 1) * N + n] : 0.0f;
  unsigned int lo = bf16_bits(f0), hi = bf16_bits(f1);
  packed[t * 256 + lane * 8 + v] = lo | (hi << 16);
}

// ---------------------------------------------------------------------------
// Kernel 1: f32 -> bf16 bulk convert (pairs -> u32 stores)
// ---------------------------------------------------------------------------
__global__ __launch_bounds__(256) void cvt_bf16_kernel(
    const float* __restrict__ in, unsigned int* __restrict__ out, int nPairs) {
  int i = blockIdx.x * blockDim.x + threadIdx.x;
  int stride = gridDim.x * blockDim.x;
  const float2* in2 = (const float2*)in;
  for (; i < nPairs; i += stride) {
    float2 v = in2[i];
    unsigned int lo = bf16_bits(v.x), hi = bf16_bits(v.y);
    out[i] = lo | (hi << 16);
  }
}

// ---------------------------------------------------------------------------
// Kernel 2: zero aggregation buffer
// ---------------------------------------------------------------------------
__global__ __launch_bounds__(256) void zero_kernel(float4* __restrict__ p, int n4) {
  int i = blockIdx.x * blockDim.x + threadIdx.x;
  int stride = gridDim.x * blockDim.x;
  float4 z = make_float4(0.f, 0.f, 0.f, 0.f);
  for (; i < n4; i += stride) p[i] = z;
}

// ---------------------------------------------------------------------------
// Kernel 3: edge message MLP + scatter-sum.
// 1 wave = 16 edges. A tiles load DIRECTLY from bf16 node state / edge feats:
//   kt 0-1 -> src row, kt 2-3 -> dst row, kt 4 -> edge feats + zero pad.
// Layer1: 40 WMMA; relu -> bf16 LDS; Layer2: 16 WMMA; 64 f32 atomics/edge.
// ---------------------------------------------------------------------------
__global__ __launch_bounds__(128) void msg_kernel(
    const unsigned short* __restrict__ hbf, const unsigned short* __restrict__ efbf,
    const int* __restrict__ esrc, const int* __restrict__ edst,
    const unsigned int* __restrict__ packed,
    const float* __restrict__ b1, const float* __restrict__ b2,
    float* __restrict__ agg) {
  __shared__ __align__(16) unsigned short hBuf[4][16 * MSG_HID];  // 16 KB

  int wave = threadIdx.x >> 5;
  int lane = threadIdx.x & 31;
  int e0   = (blockIdx.x * 4 + wave) * 16;
  int r    = lane & 15, half = lane >> 4;

  unsigned short* Hb = hBuf[wave];

  int e = e0 + r;
  int snode = esrc[e], dnode = edst[e];
  const unsigned short* srow = hbf + snode * HDIM;
  const unsigned short* drow = hbf + dnode * HDIM;

  u32x8 aT[5];
  aT[0] = loadA_bf16row(srow, 0 * 32 + half * 8);
  aT[1] = loadA_bf16row(srow, 1 * 32 + half * 8);
  aT[2] = loadA_bf16row(drow, 0 * 32 + half * 8);
  aT[3] = loadA_bf16row(drow, 1 * 32 + half * 8);
  {
    u32x8 a4 = {};
    if (half == 0) {                       // cols 128..135 = edge feats, rest pad
      uint4 x = *(const uint4*)(efbf + e * EDIM);
      a4[0] = x.x; a4[1] = x.y; a4[2] = x.z; a4[3] = x.w;
    }
    aT[4] = a4;
  }

  f32x8 zero = {};

  // ---- Layer 1: hidden = relu(A @ W1 + b1) ----
  f32x8 acc[8];
#pragma unroll
  for (int nt = 0; nt < 8; nt++) acc[nt] = zero;
#pragma unroll
  for (int kt = 0; kt < 5; kt++) {
    const unsigned int* pw = packed + (TILE_W1 + kt * 8) * 256 + lane * 8;
#pragma unroll
    for (int nt = 0; nt < 8; nt++)
      acc[nt] = wmma_bf16(aT[kt], load_tileB(pw + nt * 256), acc[nt]);
  }
#pragma unroll
  for (int nt = 0; nt < 8; nt++) {
    int n = nt * 16 + r;
    float bias = b1[n];
#pragma unroll
    for (int rr = 0; rr < 8; rr++) {
      float x = acc[nt][rr] + bias;
      x = x > 0.0f ? x : 0.0f;
      Hb[(half * 8 + rr) * MSG_HID + n] = bf16_bits(x);
    }
  }
  __syncthreads();

  // ---- Layer 2: m = hidden @ W2 + b2 ----
  f32x8 acc2[4];
#pragma unroll
  for (int nt = 0; nt < 4; nt++) acc2[nt] = zero;
#pragma unroll
  for (int kt = 0; kt < 4; kt++) {
    u32x8 a2 = loadA_bf16row(Hb + r * MSG_HID, kt * 32 + half * 8);
    const unsigned int* pw = packed + (TILE_W2 + kt * 4) * 256 + lane * 8;
#pragma unroll
    for (int nt = 0; nt < 4; nt++)
      acc2[nt] = wmma_bf16(a2, load_tileB(pw + nt * 256), acc2[nt]);
  }

  // ---- Scatter-sum to destination nodes (L2 atomics, no return) ----
#pragma unroll
  for (int nt = 0; nt < 4; nt++) {
    int n = nt * 16 + r;
    float bias = b2[n];
#pragma unroll
    for (int rr = 0; rr < 8; rr++) {
      int m = half * 8 + rr;
      int d = edst[e0 + m];
      unsafeAtomicAdd(&agg[d * HDIM + n], acc2[nt][rr] + bias);
    }
  }
}

// ---------------------------------------------------------------------------
// Kernel 4: GRU cell update. 1 wave = 16 nodes; 48 WMMAs.
// h operand loads straight from bf16 mirror; agg converts with native casts.
// Writes both f32 state (exact hold chain) and bf16 mirror.
// ---------------------------------------------------------------------------
__global__ __launch_bounds__(128) void gru_kernel(
    const float* __restrict__ holdsrc, const float* __restrict__ agg,
    const unsigned int* __restrict__ packed, const float* __restrict__ bg,
    float* __restrict__ hout, unsigned short* __restrict__ hbf, int nTiles) {
  int wave = threadIdx.x >> 5;
  int lane = threadIdx.x & 31;
  int tile = blockIdx.x * 4 + wave;
  if (tile >= nTiles) return;
  int r = lane & 15, half = lane >> 4;
  int row0 = tile * 16;

  bf16x16 ax[2];
  u32x8   ah[2];
#pragma unroll
  for (int kt = 0; kt < 2; kt++) {
    ax[kt] = loadA_f32row(agg + (row0 + r) * HDIM, kt * 32 + half * 8);
    ah[kt] = loadA_bf16row(hbf + (row0 + r) * HDIM, kt * 32 + half * 8);
  }

  f32x8 zero = {};
#pragma unroll
  for (int ft = 0; ft < 4; ft++) {
    f32x8 az = zero, ar = zero, axh = zero, ahh = zero;
#pragma unroll
    for (int kt = 0; kt < 2; kt++) {
      const unsigned int* px = packed + (TILE_WX + kt * 12) * 256 + lane * 8;
      const unsigned int* ph = packed + (TILE_WH + kt * 12) * 256 + lane * 8;
      az  = wmma_bf16v(ax[kt], load_tileB(px + ft * 256),       az);   // xz
      az  = wmma_bf16 (ah[kt], load_tileB(ph + ft * 256),       az);   // +hz
      ar  = wmma_bf16v(ax[kt], load_tileB(px + (ft + 4) * 256), ar);   // xr
      ar  = wmma_bf16 (ah[kt], load_tileB(ph + (ft + 4) * 256), ar);   // +hr
      axh = wmma_bf16v(ax[kt], load_tileB(px + (ft + 8) * 256), axh);  // xh
      ahh = wmma_bf16 (ah[kt], load_tileB(ph + (ft + 8) * 256), ahh);  // hh
    }
    int f = ft * 16 + r;
    float bz = bg[f], br = bg[HDIM + f], bh = bg[2 * HDIM + f];
#pragma unroll
    for (int rr = 0; rr < 8; rr++) {
      int m = row0 + half * 8 + rr;
      float hold = holdsrc[m * HDIM + f];
      float z  = sigmoid_f(az[rr] + bz);
      float rg = sigmoid_f(ar[rr] + br);
      float hc = tanh_f(axh[rr] + bh + rg * ahh[rr]);
      float hn = z * hold + (1.0f - z) * hc;
      hout[m * HDIM + f] = hn;
      hbf[m * HDIM + f]  = bf16_bits(hn);
    }
  }
}

// ---------------------------------------------------------------------------
// Kernel 5: readout MLP. 1 wave = 16 nodes; 16 WMMAs + shuffle reduce.
// ---------------------------------------------------------------------------
__global__ __launch_bounds__(128) void readout_kernel(
    const unsigned short* __restrict__ hbf, const unsigned int* __restrict__ packed,
    const float* __restrict__ b1, const float* __restrict__ w2,
    const float* __restrict__ b2, float* __restrict__ out, int nTiles) {
  int wave = threadIdx.x >> 5;
  int lane = threadIdx.x & 31;
  int tile = blockIdx.x * 4 + wave;
  if (tile >= nTiles) return;
  int r = lane & 15, half = lane >> 4;
  int row0 = tile * 16;

  u32x8 a[2];
#pragma unroll
  for (int kt = 0; kt < 2; kt++)
    a[kt] = loadA_bf16row(hbf + (row0 + r) * HDIM, kt * 32 + half * 8);

  f32x8 zero = {};
  f32x8 acc[8];
#pragma unroll
  for (int nt = 0; nt < 8; nt++) acc[nt] = zero;
#pragma unroll
  for (int kt = 0; kt < 2; kt++) {
    const unsigned int* pw = packed + (TILE_WR1 + kt * 8) * 256 + lane * 8;
#pragma unroll
    for (int nt = 0; nt < 8; nt++)
      acc[nt] = wmma_bf16(a[kt], load_tileB(pw + nt * 256), acc[nt]);
  }

  float part[8];
#pragma unroll
  for (int rr = 0; rr < 8; rr++) part[rr] = 0.0f;
#pragma unroll
  for (int nt = 0; nt < 8; nt++) {
    int n = nt * 16 + r;
    float bias = b1[n], w = w2[n];
#pragma unroll
    for (int rr = 0; rr < 8; rr++) {
      float x = acc[nt][rr] + bias;
      x = x > 0.0f ? x : 0.0f;
      part[rr] += x * w;
    }
  }
  float bias2 = b2[0];
#pragma unroll
  for (int rr = 0; rr < 8; rr++) {
    float p = part[rr];
    p += __shfl_xor(p, 1, 32);   // masks < 16: reduce within each 16-lane half
    p += __shfl_xor(p, 2, 32);
    p += __shfl_xor(p, 4, 32);
    p += __shfl_xor(p, 8, 32);
    if (r == 0) out[row0 + half * 8 + rr] = p + bias2;
  }
}

// ---------------------------------------------------------------------------
// Host launcher
// ---------------------------------------------------------------------------
extern "C" void kernel_launch(void* const* d_in, const int* in_sizes, int n_in,
                              void* d_out, int out_size, void* d_ws, size_t ws_size,
                              hipStream_t stream) {
  const float* node_feats = (const float*)d_in[0];
  const float* edge_feats = (const float*)d_in[1];
  const int*   edge_src   = (const int*)d_in[2];
  const int*   edge_dst   = (const int*)d_in[3];
  const float* W_msg1     = (const float*)d_in[4];
  const float* b_msg1     = (const float*)d_in[5];
  const float* W_msg2     = (const float*)d_in[6];
  const float* b_msg2     = (const float*)d_in[7];
  const float* W_x        = (const float*)d_in[8];
  const float* W_h        = (const float*)d_in[9];
  const float* b_gru      = (const float*)d_in[10];
  const float* W_r1       = (const float*)d_in[11];
  const float* b_r1       = (const float*)d_in[12];
  const float* W_r2       = (const float*)d_in[13];
  const float* b_r2       = (const float*)d_in[14];
  float* out = (float*)d_out;

  unsigned int*   packed = (unsigned int*)d_ws;
  unsigned short* efbf   = (unsigned short*)((char*)d_ws + EFBF_OFF);
  unsigned short* hbf    = (unsigned short*)((char*)d_ws + HBF_OFF);
  float*          agg    = (float*)((char*)d_ws + AGG_OFF);
  float*          hbuf   = (float*)((char*)d_ws + HBUF_OFF);

  const int nNodeTiles = N_NODES / 16;              // 3125
  const int nEdgeTiles = N_EDGES / 16;              // 25000
  const int msgBlocks  = nEdgeTiles / 4;            // 6250 (4 waves/block)
  const int nodeBlocks = (nNodeTiles + 3) / 4;      // 782
  const int aggF4      = N_NODES * HDIM / 4;        // 800000

  pack_weights_kernel<<<N_TILES_TOTAL, 256, 0, stream>>>(
      W_msg1, W_msg2, W_x, W_h, W_r1, packed);
  cvt_bf16_kernel<<<1024, 256, 0, stream>>>(
      edge_feats, (unsigned int*)efbf, N_EDGES * EDIM / 2);
  cvt_bf16_kernel<<<1024, 256, 0, stream>>>(
      node_feats, (unsigned int*)hbf, N_NODES * HDIM / 2);

  const float* holdsrc = node_feats;
  for (int t = 0; t < 2; t++) {
    zero_kernel<<<1024, 256, 0, stream>>>((float4*)agg, aggF4);
    msg_kernel<<<msgBlocks, 128, 0, stream>>>(
        hbf, efbf, edge_src, edge_dst, packed, b_msg1, b_msg2, agg);
    gru_kernel<<<nodeBlocks, 128, 0, stream>>>(
        holdsrc, agg, packed, b_gru, hbuf, hbf, nNodeTiles);
    holdsrc = hbuf;   // step 2 reads/writes hbuf per-element safely
  }

  readout_kernel<<<nodeBlocks, 128, 0, stream>>>(
      hbf, packed, b_r1, W_r2, b_r2, out, nNodeTiles);
}
